// EncoderCosineRanker_47450798686290
// MI455X (gfx1250) — matrix operand order repaired and checked by reading, not compiled, not run on hardware
//
#include <hip/hip_runtime.h>
#include <hip/hip_bf16.h>
#include <math.h>

// ---------------------------------------------------------------------------
// Model dims (compile-time constants from the reference)
// ---------------------------------------------------------------------------
#define BB_    2
#define SS_    4096
#define DD_    768
#define HH_    12
#define DH_    64
#define WW_    128
#define CC_    32          // S / W
#define D4_    3072
#define KCLS_  128
#define NTOK_  (BB_ * SS_)   // 8192
#define NEGV   (-1.0e9f)

typedef __attribute__((ext_vector_type(16))) _Float16 v16h;
typedef __attribute__((ext_vector_type(8)))  float    v8f;
typedef __attribute__((ext_vector_type(4)))  unsigned int u32x4;

union FragA { v16h v; u32x4 u[2]; };   // 32-byte WMMA fragment

__device__ __forceinline__ float waveMax(float v) {
    for (int o = 16; o > 0; o >>= 1) v = fmaxf(v, __shfl_xor(v, o));
    return v;
}
__device__ __forceinline__ float waveSum(float v) {
    for (int o = 16; o > 0; o >>= 1) v += __shfl_xor(v, o);
    return v;
}

// ---------------------------------------------------------------------------
// Weight convert + transpose: wt[n][k] = (f16) w[k][n]   (K-contiguous rows)
// ---------------------------------------------------------------------------
__global__ void cvt_t_kernel(const float* __restrict__ w, _Float16* __restrict__ wt,
                             int K, int N) {
    size_t i = (size_t)blockIdx.x * 256 + threadIdx.x;
    if (i >= (size_t)K * N) return;
    int n  = (int)(i / K);
    int kk = (int)(i % K);
    wt[i] = (_Float16)w[(size_t)kk * N + n];
}

// ---------------------------------------------------------------------------
// x = LN(emb[sent] + pos); writes f32 (residual path) and f16 (WMMA path)
// ---------------------------------------------------------------------------
__global__ void embed_ln_kernel(const int* __restrict__ sent, const float* __restrict__ emb,
                                const float* __restrict__ pos, const float* __restrict__ g,
                                const float* __restrict__ bta, float* __restrict__ xo,
                                _Float16* __restrict__ xh) {
    __shared__ float wsum[8], wsq[8], stat[2];
    int t = blockIdx.x;                 // token index in [0, NTOK)
    int s = t & (SS_ - 1);
    const float* er = emb + (size_t)sent[t] * DD_;
    const float* pr = pos + (size_t)s * DD_;
    float vals[3], ls = 0.f, lq = 0.f;
#pragma unroll
    for (int i = 0; i < 3; ++i) {
        int e = threadIdx.x + i * 256;
        float v = er[e] + pr[e];
        vals[i] = v; ls += v; lq += v * v;
    }
    ls = waveSum(ls); lq = waveSum(lq);
    if ((threadIdx.x & 31) == 0) { wsum[threadIdx.x >> 5] = ls; wsq[threadIdx.x >> 5] = lq; }
    __syncthreads();
    if (threadIdx.x == 0) {
        float a = 0.f, b = 0.f;
        for (int i = 0; i < 8; ++i) { a += wsum[i]; b += wsq[i]; }
        float mean = a / (float)DD_;
        stat[0] = mean;
        stat[1] = rsqrtf(b / (float)DD_ - mean * mean + 1e-5f);
    }
    __syncthreads();
    float mean = stat[0], inv = stat[1];
#pragma unroll
    for (int i = 0; i < 3; ++i) {
        int e = threadIdx.x + i * 256;
        float v = (vals[i] - mean) * inv * g[e] + bta[e];
        xo[(size_t)t * DD_ + e] = v;
        xh[(size_t)t * DD_ + e] = (_Float16)v;
    }
}

// ---------------------------------------------------------------------------
// out = LN(x + y); optional f16 copy.
// ---------------------------------------------------------------------------
__global__ void add_ln_kernel(const float* __restrict__ x, const float* __restrict__ y,
                              const float* __restrict__ g, const float* __restrict__ bta,
                              float* __restrict__ xo, _Float16* __restrict__ xh) {
    __shared__ float wsum[8], wsq[8], stat[2];
    int t = blockIdx.x;
    float vals[3], ls = 0.f, lq = 0.f;
#pragma unroll
    for (int i = 0; i < 3; ++i) {
        int e = threadIdx.x + i * 256;
        float v = x[(size_t)t * DD_ + e] + y[(size_t)t * DD_ + e];
        vals[i] = v; ls += v; lq += v * v;
    }
    ls = waveSum(ls); lq = waveSum(lq);
    if ((threadIdx.x & 31) == 0) { wsum[threadIdx.x >> 5] = ls; wsq[threadIdx.x >> 5] = lq; }
    __syncthreads();
    if (threadIdx.x == 0) {
        float a = 0.f, b = 0.f;
        for (int i = 0; i < 8; ++i) { a += wsum[i]; b += wsq[i]; }
        float mean = a / (float)DD_;
        stat[0] = mean;
        stat[1] = rsqrtf(b / (float)DD_ - mean * mean + 1e-5f);
    }
    __syncthreads();
    float mean = stat[0], inv = stat[1];
#pragma unroll
    for (int i = 0; i < 3; ++i) {
        int e = threadIdx.x + i * 256;
        float v = (vals[i] - mean) * inv * g[e] + bta[e];
        xo[(size_t)t * DD_ + e] = v;
        if (xh) xh[(size_t)t * DD_ + e] = (_Float16)v;
    }
}

// ---------------------------------------------------------------------------
// WMMA GEMM: out[M,N] = A[M,K](f16) @ Bt[N,K]^T (f16) + bias
// block = 256 thr (8 waves), each wave -> 16x64 output tile; block -> 128x64.
// grid = (N/64, M/128).  ACT==1 -> exact GELU.
// perm!=0 -> f16 output scattered to [b,h,s,dh] head-split layout.
// ---------------------------------------------------------------------------
template <int ACT>
__global__ void gemm_wmma_kernel(const _Float16* __restrict__ A,
                                 const _Float16* __restrict__ Bt,
                                 const float* __restrict__ bias,
                                 float* __restrict__ outF,
                                 _Float16* __restrict__ outH,
                                 int M, int N, int K, int perm) {
    const int lane  = threadIdx.x & 31;
    const int wid   = threadIdx.x >> 5;
    const int lrow  = lane & 15;
    const int lhalf = lane >> 4;
    const int m0 = blockIdx.y * 128 + wid * 16;
    const int n0 = blockIdx.x * 64;

    v8f zero = {0.f,0.f,0.f,0.f,0.f,0.f,0.f,0.f};
    v8f acc[4] = {zero, zero, zero, zero};

    const _Float16* aRow = A + (size_t)(m0 + lrow) * K;
#pragma unroll 2
    for (int k0 = 0; k0 < K; k0 += 32) {
        if (k0 + 256 < K) __builtin_prefetch(aRow + k0 + 256, 0, 0);
        FragA a;
        {   // contiguous k order used here, B the same: lane<16 -> K {0..15}; lane>=16 -> K {16..31}
            const u32x4* gp = (const u32x4*)(aRow + k0 + lhalf * 16);
            a.u[0] = gp[0]; a.u[1] = gp[1];
        }
#pragma unroll
        for (int j = 0; j < 4; ++j) {
            FragA b;
            const u32x4* gp = (const u32x4*)(Bt + (size_t)(n0 + j * 16 + lrow) * K + k0 + lhalf * 16);
            b.u[0] = gp[0]; b.u[1] = gp[1];
            acc[j] = __builtin_amdgcn_wmma_f32_16x16x32_f16(
                         false, a.v, false, b.v, (short)0, acc[j], false, false);
        }
    }
    float bj[4];
#pragma unroll
    for (int j = 0; j < 4; ++j) bj[j] = bias[n0 + j * 16 + lrow];
#pragma unroll
    for (int j = 0; j < 4; ++j) {
#pragma unroll
        for (int r = 0; r < 8; ++r) {
            int m = m0 + r + lhalf * 8;
            int n = n0 + j * 16 + lrow;
            float v = acc[j][r] + bj[j];
            if (ACT == 1) v = 0.5f * v * (1.0f + erff(v * 0.70710678118654752f));
            if (outF) outF[(size_t)m * N + n] = v;
            if (outH) {
                size_t idx;
                if (perm) {      // m = b*S+s ; n = h*64+d -> [b,h,s,d]
                    int b = m >> 12, s = m & (SS_ - 1);
                    int h = n >> 6,  d = n & 63;
                    idx = (((size_t)(b * HH_ + h) * SS_) + s) * DH_ + d;
                } else {
                    idx = (size_t)m * N + n;
                }
                outH[idx] = (_Float16)v;
            }
        }
    }
}

// ---------------------------------------------------------------------------
// Band scores: per (b,h,c) block (256 thr). sb[b,h,s,384] (masked, scaled).
// Q @ K^T via WMMA. Key-validity mask is per-lane-column invariant; the sent
// load is unconditional (clamped index) so no divergent exec-mask branches.
// ---------------------------------------------------------------------------
__global__ void band_scores_kernel(const _Float16* __restrict__ q,
                                   const _Float16* __restrict__ k,
                                   const int* __restrict__ sent,
                                   float* __restrict__ sb, float scale) {
    int blk = blockIdx.x;
    int c = blk & (CC_ - 1);
    int h = (blk >> 5) % HH_;
    int b = blk / (CC_ * HH_);
    int lane = threadIdx.x & 31, wid = threadIdx.x >> 5;
    int lrow = lane & 15, lhalf = lane >> 4;
    size_t bh = (size_t)(b * HH_ + h) * SS_;

    FragA a0, a1;   // two K-steps over DH=64, reused for all 24 key tiles
    {
        const _Float16* qr = q + (bh + c * WW_ + wid * 16 + lrow) * DH_;
        const u32x4* g0 = (const u32x4*)(qr + lhalf * 16);
        a0.u[0] = g0[0]; a0.u[1] = g0[1];
        const u32x4* g1 = (const u32x4*)(qr + 32 + lhalf * 16);
        a1.u[0] = g1[0]; a1.u[1] = g1[1];
    }
    const int qibase = wid * 16 + lhalf * 8;
    float* srow0 = sb + (bh + c * WW_) * 384;
#pragma unroll 2
    for (int nt = 0; nt < 24; ++nt) {
        int kcol = nt * 16 + lrow;
        int kpos = (c - 1) * WW_ + kcol;
        int posc = min(max(kpos, 0), SS_ - 1);
        const _Float16* kr = k + (bh + posc) * DH_;
        FragA b0, b1;
        { const u32x4* g = (const u32x4*)(kr + lhalf * 16);      b0.u[0] = g[0]; b0.u[1] = g[1]; }
        { const u32x4* g = (const u32x4*)(kr + 32 + lhalf * 16); b1.u[0] = g[0]; b1.u[1] = g[1]; }
        // unconditional mask load (index already clamped) -> no divergence
        int sv = sent[b * SS_ + posc];
        v8f acc = {0.f,0.f,0.f,0.f,0.f,0.f,0.f,0.f};
        acc = __builtin_amdgcn_wmma_f32_16x16x32_f16(false, a0.v, false, b0.v, (short)0, acc, false, false);
        acc = __builtin_amdgcn_wmma_f32_16x16x32_f16(false, a1.v, false, b1.v, (short)0, acc, false, false);
        bool keyok = (kpos == posc) & (sv != 1);
        int  dcol  = kcol - WW_ - qibase;      // band dist for r=0
#pragma unroll
        for (int r = 0; r < 8; ++r) {
            bool band = (unsigned)(dcol - r + WW_) <= (unsigned)(2 * WW_); // |dcol-r|<=W
            float v = (keyok & band) ? acc[r] * scale : NEGV;
            srow0[(size_t)(qibase + r) * 384 + kcol] = v;
        }
    }
}

// ---------------------------------------------------------------------------
// Global-token scores sg[b,h,s,2] = q[s] . k[gidx]  (unmasked, scaled)
// ---------------------------------------------------------------------------
__global__ void global_scores_kernel(const _Float16* __restrict__ q,
                                     const _Float16* __restrict__ k,
                                     const int* __restrict__ sp, const int* __restrict__ ep,
                                     float* __restrict__ sg, float scale) {
    size_t i = (size_t)blockIdx.x * 256 + threadIdx.x;   // b*H*S rows, exact
    int s = (int)(i & (SS_ - 1));
    int h = (int)((i >> 12) % HH_);
    int b = (int)(i / ((size_t)SS_ * HH_));
    size_t bh = (size_t)(b * HH_ + h) * SS_;
    const _Float16* qr = q + (bh + s) * DH_;
#pragma unroll
    for (int g = 0; g < 2; ++g) {
        int gi = (g == 0) ? sp[b] : ep[b];
        const _Float16* kr = k + (bh + gi) * DH_;
        float dot = 0.f;
        for (int d = 0; d < DH_; ++d) dot += (float)qr[d] * (float)kr[d];
        sg[i * 2 + g] = dot * scale;
    }
}

// ---------------------------------------------------------------------------
// Row softmax over 386 = 384 band + 2 global. 1 wave per row, 8 rows/block.
// ---------------------------------------------------------------------------
__global__ void softmax_kernel(const float* __restrict__ sb, const float* __restrict__ sg,
                               _Float16* __restrict__ pb, float* __restrict__ pg) {
    int lane = threadIdx.x & 31, wid = threadIdx.x >> 5;
    size_t row = (size_t)blockIdx.x * 8 + wid;
    const float* srow = sb + row * 384;
    float v[12], m = -3.4e38f;
#pragma unroll
    for (int i = 0; i < 12; ++i) { v[i] = srow[i * 32 + lane]; m = fmaxf(m, v[i]); }
    float gs = (lane < 2) ? sg[row * 2 + lane] : -3.4e38f;
    m = fmaxf(m, gs);
    m = waveMax(m);
    float ssum = 0.f;
#pragma unroll
    for (int i = 0; i < 12; ++i) { v[i] = __expf(v[i] - m); ssum += v[i]; }
    float ge = (lane < 2) ? __expf(gs - m) : 0.f;
    ssum += ge;
    ssum = waveSum(ssum);
    float inv = 1.0f / ssum;
    _Float16* prow = pb + row * 384;
#pragma unroll
    for (int i = 0; i < 12; ++i) prow[i * 32 + lane] = (_Float16)(v[i] * inv);
    if (lane < 2) pg[row * 2 + lane] = ge * inv;
}

// ---------------------------------------------------------------------------
// Band output: o = Pband @ Vwindow (WMMA, V transposed through LDS) + Pg @ gv.
// One block per (b,h,c); writes o as f16 in [b,s,h*64+d] layout.
// ---------------------------------------------------------------------------
__global__ void band_out_kernel(const _Float16* __restrict__ pb, const float* __restrict__ pg,
                                const _Float16* __restrict__ vmat,
                                const int* __restrict__ sp, const int* __restrict__ ep,
                                _Float16* __restrict__ oh) {
    __shared__ alignas(16) _Float16 vT[DH_ * 392];     // 64 x (384 + pad)
    int blk = blockIdx.x;
    int c = blk & (CC_ - 1);
    int h = (blk >> 5) % HH_;
    int b = blk / (CC_ * HH_);
    size_t bh = (size_t)(b * HH_ + h) * SS_;

    // stage V window transposed (vT[d][kk]), zero-fill out-of-range rows
    for (int e = threadIdx.x; e < 384 * DH_; e += 256) {
        int kk = e >> 6, d = e & 63;
        int pos = (c - 1) * WW_ + kk;
        _Float16 val = (_Float16)0.0f;
        if (pos >= 0 && pos < SS_) val = vmat[(bh + pos) * DH_ + d];
        vT[d * 392 + kk] = val;
    }
    __syncthreads();

    int lane = threadIdx.x & 31, wid = threadIdx.x >> 5;
    int lrow = lane & 15, lhalf = lane >> 4;
    v8f zero = {0.f,0.f,0.f,0.f,0.f,0.f,0.f,0.f};
    v8f acc[4] = {zero, zero, zero, zero};
    const _Float16* prow = pb + (bh + c * WW_ + wid * 16 + lrow) * 384;
#pragma unroll 2
    for (int kk0 = 0; kk0 < 384; kk0 += 32) {
        FragA a;
        const u32x4* g = (const u32x4*)(prow + kk0 + lhalf * 16);
        a.u[0] = g[0]; a.u[1] = g[1];
#pragma unroll
        for (int j = 0; j < 4; ++j) {
            int d = j * 16 + lrow;
            FragA bf;
            const u32x4* lp = (const u32x4*)(&vT[d * 392 + kk0 + lhalf * 16]);
            bf.u[0] = lp[0]; bf.u[1] = lp[1];
            acc[j] = __builtin_amdgcn_wmma_f32_16x16x32_f16(
                         false, a.v, false, bf.v, (short)0, acc[j], false, false);
        }
    }
    // global-token contribution: gather gv once per j, pg once per r
    int gi0 = sp[b], gi1 = ep[b];
    float gv0[4], gv1[4];
#pragma unroll
    for (int j = 0; j < 4; ++j) {
        int d = j * 16 + lrow;
        gv0[j] = (float)vmat[(bh + gi0) * DH_ + d];
        gv1[j] = (float)vmat[(bh + gi1) * DH_ + d];
    }
    const int qibase = wid * 16 + lhalf * 8;
#pragma unroll
    for (int r = 0; r < 8; ++r) {
        int s = c * WW_ + qibase + r;
        float p0 = pg[(bh + s) * 2 + 0], p1 = pg[(bh + s) * 2 + 1];
        _Float16* orow = oh + ((size_t)b * SS_ + s) * DD_ + h * DH_;
#pragma unroll
        for (int j = 0; j < 4; ++j) {
            float val = acc[j][r] + p0 * gv0[j] + p1 * gv1[j];
            orow[j * 16 + lrow] = (_Float16)val;
        }
    }
}

// ---------------------------------------------------------------------------
// Global-token attention rows: og = softmax(mask(q_g . K^T)) @ V,
// overwrites o rows at gidx. One block per (b,h); serial over g (last wins).
// ---------------------------------------------------------------------------
__global__ void global_attn_kernel(const _Float16* __restrict__ q, const _Float16* __restrict__ k,
                                   const _Float16* __restrict__ vmat, const int* __restrict__ sent,
                                   const int* __restrict__ sp, const int* __restrict__ ep,
                                   _Float16* __restrict__ oh, float scale) {
    __shared__ float sc[SS_];
    __shared__ float qs[DH_];
    __shared__ float red[DH_];
    __shared__ float wred[8];
    __shared__ float stat[2];
    int b = blockIdx.x / HH_, h = blockIdx.x % HH_;
    size_t bh = (size_t)(b * HH_ + h) * SS_;
    for (int g = 0; g < 2; ++g) {
        int gi = (g == 0) ? sp[b] : ep[b];
        if (threadIdx.x < DH_) {
            qs[threadIdx.x]  = (float)q[(bh + gi) * DH_ + threadIdx.x];
            red[threadIdx.x] = 0.f;
        }
        __syncthreads();
        float lmax = -3.4e38f;
        for (int s = threadIdx.x; s < SS_; s += 256) {
            const _Float16* kr = k + (bh + s) * DH_;
            float dot = 0.f;
            for (int d = 0; d < DH_; ++d) dot += qs[d] * (float)kr[d];
            float v = (sent[b * SS_ + s] != 1) ? dot * scale : NEGV;
            sc[s] = v;
            lmax = fmaxf(lmax, v);
        }
        lmax = waveMax(lmax);
        if ((threadIdx.x & 31) == 0) wred[threadIdx.x >> 5] = lmax;
        __syncthreads();
        if (threadIdx.x == 0) {
            float m = wred[0];
            for (int i = 1; i < 8; ++i) m = fmaxf(m, wred[i]);
            stat[0] = m;
        }
        __syncthreads();
        float m = stat[0], lsum = 0.f;
        for (int s = threadIdx.x; s < SS_; s += 256) {
            float e = __expf(sc[s] - m);
            sc[s] = e; lsum += e;
        }
        lsum = waveSum(lsum);
        if ((threadIdx.x & 31) == 0) wred[threadIdx.x >> 5] = lsum;
        __syncthreads();
        if (threadIdx.x == 0) {
            float t = 0.f;
            for (int i = 0; i < 8; ++i) t += wred[i];
            stat[1] = 1.0f / t;
        }
        __syncthreads();
        float inv = stat[1];
        float od[DH_];
        for (int d = 0; d < DH_; ++d) od[d] = 0.f;
        for (int s = threadIdx.x; s < SS_; s += 256) {
            float p = sc[s] * inv;
            const _Float16* vr = vmat + (bh + s) * DH_;
            for (int d = 0; d < DH_; ++d) od[d] += p * (float)vr[d];
        }
        for (int d = 0; d < DH_; ++d) atomicAdd(&red[d], od[d]);   // ds_add_f32
        __syncthreads();
        if (threadIdx.x < DH_)
            oh[((size_t)b * SS_ + gi) * DD_ + h * DH_ + threadIdx.x] = (_Float16)red[threadIdx.x];
        __syncthreads();
    }
}

// ---------------------------------------------------------------------------
// mention = [h[sp]; h[ep]] ; logits = mention @ ex^T ; loss = NLL. One block.
// out[0..255] = logits (B x K), out[256] = loss.
// ---------------------------------------------------------------------------
__global__ void mention_kernel(const float* __restrict__ hf, const float* __restrict__ ex,
                               const int* __restrict__ sp, const int* __restrict__ ep,
                               const int* __restrict__ labels, float* __restrict__ out) {
    __shared__ float lg[BB_ * KCLS_];
    __shared__ float lossp[BB_];
    int t = threadIdx.x;
    int b = t >> 7, kk = t & (KCLS_ - 1);
    const float* h0 = hf + ((size_t)b * SS_ + sp[b]) * DD_;
    const float* h1 = hf + ((size_t)b * SS_ + ep[b]) * DD_;
    const float* er = ex + (size_t)kk * (2 * DD_);
    float acc = 0.f;
    for (int j = 0; j < DD_; ++j) acc += h0[j] * er[j];
    for (int j = 0; j < DD_; ++j) acc += h1[j] * er[DD_ + j];
    lg[t]  = acc;
    out[t] = acc;
    __syncthreads();
    if (t < BB_) {
        const float* row = lg + t * KCLS_;
        float m = row[0];
        for (int i = 1; i < KCLS_; ++i) m = fmaxf(m, row[i]);
        float sum = 0.f;
        for (int i = 0; i < KCLS_; ++i) sum += __expf(row[i] - m);
        lossp[t] = -(row[labels[t]] - m - logf(sum));
    }
    __syncthreads();
    if (t == 0) out[BB_ * KCLS_] = lossp[0] + lossp[1];
}

// ---------------------------------------------------------------------------
// Host launcher
// ---------------------------------------------------------------------------
extern "C" void kernel_launch(void* const* d_in, const int* in_sizes, int n_in,
                              void* d_out, int out_size, void* d_ws, size_t ws_size,
                              hipStream_t stream) {
    (void)in_sizes; (void)n_in; (void)out_size; (void)ws_size;
    const int*   sent = (const int*)d_in[0];
    const int*   sp   = (const int*)d_in[1];
    const int*   ep   = (const int*)d_in[2];
    const int*   lab  = (const int*)d_in[3];
    const float* emb  = (const float*)d_in[4];
    const float* pos  = (const float*)d_in[5];
    const float* ln0g = (const float*)d_in[6];  const float* ln0b = (const float*)d_in[7];
    const float* wq   = (const float*)d_in[8];  const float* bq   = (const float*)d_in[9];
    const float* wk   = (const float*)d_in[10]; const float* bk   = (const float*)d_in[11];
    const float* wv   = (const float*)d_in[12]; const float* bv   = (const float*)d_in[13];
    const float* wo   = (const float*)d_in[14]; const float* bo   = (const float*)d_in[15];
    const float* ln1g = (const float*)d_in[16]; const float* ln1b = (const float*)d_in[17];
    const float* w1   = (const float*)d_in[18]; const float* b1   = (const float*)d_in[19];
    const float* w2   = (const float*)d_in[20]; const float* b2   = (const float*)d_in[21];
    const float* ln2g = (const float*)d_in[22]; const float* ln2b = (const float*)d_in[23];
    const float* ex   = (const float*)d_in[24];

    size_t off = 0;
    auto alloc = [&](size_t bytes) -> char* {
        char* p = (char*)d_ws + off;
        off += (bytes + 255) & ~(size_t)255;
        return p;
    };
    const size_t NT = NTOK_;
    float*     xf   = (float*)    alloc(NT * DD_ * 4);
    _Float16*  xh   = (_Float16*) alloc(NT * DD_ * 2);
    _Float16*  wqT  = (_Float16*) alloc((size_t)DD_ * DD_ * 2);
    _Float16*  wkT  = (_Float16*) alloc((size_t)DD_ * DD_ * 2);
    _Float16*  wvT  = (_Float16*) alloc((size_t)DD_ * DD_ * 2);
    _Float16*  woT  = (_Float16*) alloc((size_t)DD_ * DD_ * 2);
    _Float16*  w1T  = (_Float16*) alloc((size_t)DD_ * D4_ * 2);
    _Float16*  w2T  = (_Float16*) alloc((size_t)DD_ * D4_ * 2);
    _Float16*  qh   = (_Float16*) alloc(NT * DD_ * 2);
    _Float16*  kh   = (_Float16*) alloc(NT * DD_ * 2);
    _Float16*  vh   = (_Float16*) alloc(NT * DD_ * 2);
    float*     sb   = (float*)    alloc((size_t)BB_ * HH_ * SS_ * 384 * 4);
    float*     sg   = (float*)    alloc((size_t)BB_ * HH_ * SS_ * 2 * 4);
    _Float16*  pb   = (_Float16*) alloc((size_t)BB_ * HH_ * SS_ * 384 * 2);
    float*     pg   = (float*)    alloc((size_t)BB_ * HH_ * SS_ * 2 * 4);
    _Float16*  oh   = (_Float16*) alloc(NT * DD_ * 2);
    float*     proj = (float*)    alloc(NT * DD_ * 4);
    float*     x1f  = (float*)    alloc(NT * DD_ * 4);
    _Float16*  x1h  = (_Float16*) alloc(NT * DD_ * 2);
    _Float16*  ff1h = (_Float16*) alloc(NT * D4_ * 2);
    float*     ff2f = (float*)    alloc(NT * DD_ * 4);
    float*     hfin = (float*)    alloc(NT * DD_ * 4);

    const float scale = 0.125f;   // 1/sqrt(64)
    dim3 blk(256);

    // weight transpose + f32->f16
    cvt_t_kernel<<<(DD_ * DD_ + 255) / 256, blk, 0, stream>>>(wq, wqT, DD_, DD_);
    cvt_t_kernel<<<(DD_ * DD_ + 255) / 256, blk, 0, stream>>>(wk, wkT, DD_, DD_);
    cvt_t_kernel<<<(DD_ * DD_ + 255) / 256, blk, 0, stream>>>(wv, wvT, DD_, DD_);
    cvt_t_kernel<<<(DD_ * DD_ + 255) / 256, blk, 0, stream>>>(wo, woT, DD_, DD_);
    cvt_t_kernel<<<(DD_ * D4_ + 255) / 256, blk, 0, stream>>>(w1, w1T, DD_, D4_);
    cvt_t_kernel<<<(DD_ * D4_ + 255) / 256, blk, 0, stream>>>(w2, w2T, D4_, DD_);

    // embed + LN0
    embed_ln_kernel<<<NTOK_, blk, 0, stream>>>(sent, emb, pos, ln0g, ln0b, xf, xh);

    // QKV projections (WMMA), head-split f16 outputs
    gemm_wmma_kernel<0><<<dim3(DD_ / 64, NTOK_ / 128), blk, 0, stream>>>(
        xh, wqT, bq, nullptr, qh, NTOK_, DD_, DD_, 1);
    gemm_wmma_kernel<0><<<dim3(DD_ / 64, NTOK_ / 128), blk, 0, stream>>>(
        xh, wkT, bk, nullptr, kh, NTOK_, DD_, DD_, 1);
    gemm_wmma_kernel<0><<<dim3(DD_ / 64, NTOK_ / 128), blk, 0, stream>>>(
        xh, wvT, bv, nullptr, vh, NTOK_, DD_, DD_, 1);

    // attention scores + softmax
    band_scores_kernel<<<BB_ * HH_ * CC_, blk, 0, stream>>>(qh, kh, sent, sb, scale);
    global_scores_kernel<<<(BB_ * HH_ * SS_) / 256, blk, 0, stream>>>(qh, kh, sp, ep, sg, scale);
    softmax_kernel<<<(BB_ * HH_ * SS_) / 8, blk, 0, stream>>>(sb, sg, pb, pg);

    // attention outputs
    band_out_kernel<<<BB_ * HH_ * CC_, blk, 0, stream>>>(pb, pg, vh, sp, ep, oh);
    global_attn_kernel<<<BB_ * HH_, blk, 0, stream>>>(qh, kh, vh, sent, sp, ep, oh, scale);

    // output projection + residual LN1
    gemm_wmma_kernel<0><<<dim3(DD_ / 64, NTOK_ / 128), blk, 0, stream>>>(
        oh, woT, bo, proj, nullptr, NTOK_, DD_, DD_, 0);
    add_ln_kernel<<<NTOK_, blk, 0, stream>>>(xf, proj, ln1g, ln1b, x1f, x1h);

    // FFN
    gemm_wmma_kernel<1><<<dim3(D4_ / 64, NTOK_ / 128), blk, 0, stream>>>(
        x1h, w1T, b1, nullptr, ff1h, NTOK_, D4_, DD_, 0);
    gemm_wmma_kernel<0><<<dim3(DD_ / 64, NTOK_ / 128), blk, 0, stream>>>(
        ff1h, w2T, b2, ff2f, nullptr, NTOK_, DD_, D4_, 0);
    add_ln_kernel<<<NTOK_, blk, 0, stream>>>(x1f, ff2f, ln2g, ln2b, hfin, nullptr);

    // mention -> logits -> loss
    mention_kernel<<<1, blk, 0, stream>>>(hfin, ex, sp, ep, lab, (float*)d_out);
}
